// _GQAAttention_30992484008191
// MI455X (gfx1250) — compile-verified, matrix operand order checked by reading, not run on hardware
//
#include <hip/hip_runtime.h>
#include <hip/hip_bf16.h>

typedef __bf16 bf16;
typedef __attribute__((ext_vector_type(16))) __bf16 v16bf;
typedef __attribute__((ext_vector_type(8)))  __bf16 v8bf;
typedef __attribute__((ext_vector_type(8)))  float  v8f;

#define B_   2
#define L_   2048
#define D_   2048
#define H_   32
#define HKV_ 8
#define HD_  64
#define NREP_ 4
#define KVD_ 512
#define BL_  4096
#define NEGB (-1.0e9f)

__device__ __forceinline__ v8f wmma_bf16(v16bf a, v16bf b, v8f c) {
  return __builtin_amdgcn_wmma_f32_16x16x32_bf16(false, a, false, b, (short)0, c, false, false);
}

// Per ISA 16-bit A layout: lane half 'hi' holds K = 8*hi + [0..7] (elems 0-7)
// and K = 16 + 8*hi + [0..7] (elems 8-15), row-contiguous tiles in LDS.
__device__ __forceinline__ v16bf frag_ld(const bf16* rowbase, int koff, int hi) {
  v8bf lo = *(const v8bf*)(rowbase + koff + 8 * hi);
  v8bf hc = *(const v8bf*)(rowbase + koff + 16 + 8 * hi);
  v16bf r;
#pragma unroll
  for (int i = 0; i < 8; i++) { r[i] = lo[i]; r[i + 8] = hc[i]; }
  return r;
}

// gfx1250 async global->LDS copy (VFLAT encoding, tracked by ASYNCcnt).
// Low 32 bits of a generic LDS address are the LDS byte offset.
__device__ __forceinline__ void async_copy_b128(const bf16* gsrc, bf16* lds_dst) {
  unsigned lds_off = (unsigned)(size_t)lds_dst;
  asm volatile("global_load_async_to_lds_b128 %0, %1, off"
               :: "v"(lds_off), "v"(gsrc) : "memory");
}
__device__ __forceinline__ void async_wait0() {
  asm volatile("s_wait_asynccnt 0x0" ::: "memory");
}

// ---------------- fp32 -> bf16 convert ----------------
__global__ void f32_to_bf16_k(const float* __restrict__ in, bf16* __restrict__ out, size_t n) {
  size_t i = (size_t)blockIdx.x * blockDim.x + threadIdx.x;
  size_t stride = (size_t)gridDim.x * blockDim.x;
  for (; i < n; i += stride) out[i] = (bf16)in[i];
}

// ---------------- RoPE (rotate-half, in place on bf16) ----------------
__global__ void rope_k(bf16* __restrict__ buf, int nh, int W, int total) {
  int idx = blockIdx.x * blockDim.x + threadIdx.x;
  if (idx >= total) return;
  int i    = idx & 31;
  int t    = idx >> 5;
  int head = t % nh;
  int row  = t / nh;            // b*L + l
  int l    = row & (L_ - 1);
  float ang = (float)l * __powf(10000.0f, -(float)i * (1.0f / 32.0f));
  float s, c;
  __sincosf(ang, &s, &c);
  size_t base = (size_t)row * W + head * HD_ + i;
  float a = (float)buf[base], b = (float)buf[base + 32];
  buf[base]      = (bf16)(a * c - b * s);
  buf[base + 32] = (bf16)(b * c + a * s);
}

// ---------------- WMMA GEMM: C[M,N] = A[M,K] @ Bw[K,N] (+bias) ----------------
// Block tile 256x64, 8 waves, each wave owns a 32x64 strip (2x4 accumulators).
// Double-buffered LDS: async DMA of tile i+1 overlaps WMMA on tile i.
#define GM_MT 256
#define GM_NT 64
#define GM_KT 32

__global__ __launch_bounds__(256) void gemm_wmma_k(
    const bf16* __restrict__ A, const bf16* __restrict__ Bw,
    const float* __restrict__ bias, bf16* __restrict__ Cb, float* __restrict__ Cf,
    int M, int N, int K)
{
  __shared__ __align__(16) bf16 As[2][GM_MT * GM_KT];  // [256][32] row-major (k contiguous)
  __shared__ __align__(16) bf16 Bt[2][GM_NT * GM_KT];  // [64][32]  B^T (k contiguous)
  const int tid  = threadIdx.x;
  const int lane = tid & 31, wave = tid >> 5;
  const int hi = lane >> 4, lr = lane & 15;
  const int m0 = blockIdx.y * GM_MT, n0 = blockIdx.x * GM_NT;
  const int bkk = tid >> 3, bcc = tid & 7;             // B staging coords

  v8f acc[2][4] = {{v8f{}, v8f{}, v8f{}, v8f{}}, {v8f{}, v8f{}, v8f{}, v8f{}}};

  // ---- prologue: stage tile 0 ----
#pragma unroll
  for (int i = 0; i < 4; i++) {
    int c = tid + i * 256;                 // 0..1023
    int row = c >> 2, cc = c & 3;
    async_copy_b128(&A[(size_t)(m0 + row) * K + cc * 8], &As[0][row * GM_KT + cc * 8]);
  }
  {
    v8bf tv = *(const v8bf*)&Bw[(size_t)bkk * N + n0 + bcc * 8];
#pragma unroll
    for (int e = 0; e < 8; e++) Bt[0][(bcc * 8 + e) * GM_KT + bkk] = tv[e];
  }
  async_wait0();
  __syncthreads();

  int buf = 0;
  for (int k0 = 0; k0 < K; k0 += GM_KT, buf ^= 1) {
    const int kn = k0 + GM_KT;
    const bool have_next = kn < K;
    v8bf tvn;
    if (have_next) {
      // issue async DMA of next A tile into the other buffer (overlaps WMMAs)
#pragma unroll
      for (int i = 0; i < 4; i++) {
        int c = tid + i * 256;
        int row = c >> 2, cc = c & 3;
        async_copy_b128(&A[(size_t)(m0 + row) * K + kn + cc * 8],
                        &As[buf ^ 1][row * GM_KT + cc * 8]);
      }
      // start next B-tile global load; scatter to LDS after compute
      tvn = *(const v8bf*)&Bw[(size_t)(kn + bkk) * N + n0 + bcc * 8];
      if (kn + GM_KT < K)
        __builtin_prefetch(&Bw[(size_t)(kn + GM_KT + bkk) * N + n0], 0, 1);
    }

    // ---- compute on current buffer ----
    v16bf af0 = frag_ld(&As[buf][(wave * 32 + lr) * GM_KT], 0, hi);
    v16bf af1 = frag_ld(&As[buf][(wave * 32 + 16 + lr) * GM_KT], 0, hi);
#pragma unroll
    for (int c = 0; c < 4; c++) {
      v16bf bfr = frag_ld(&Bt[buf][(c * 16 + lr) * GM_KT], 0, hi);
      acc[0][c] = wmma_bf16(af0, bfr, acc[0][c]);
      acc[1][c] = wmma_bf16(af1, bfr, acc[1][c]);
    }

    if (have_next) {
#pragma unroll
      for (int e = 0; e < 8; e++) Bt[buf ^ 1][(bcc * 8 + e) * GM_KT + bkk] = tvn[e];
    }
    async_wait0();        // publish next A tile
    __syncthreads();      // (also orders Bt scatter via dscnt wait)
  }

  // Epilogue. D layout: elem (half, c, r) -> row m = 32*wave + 16*half + r + 8*hi,
  // col n = 16*c + lr. Bias depends only on n -> hoisted.
#pragma unroll
  for (int c = 0; c < 4; c++) {
    int n = n0 + c * 16 + lr;
    float bv = bias ? bias[n] : 0.0f;
#pragma unroll
    for (int half = 0; half < 2; half++) {
#pragma unroll
      for (int r = 0; r < 8; r++) {
        int m = m0 + wave * 32 + half * 16 + r + 8 * hi;
        float v = acc[half][c][r] + bv;
        if (Cb) Cb[(size_t)m * N + n] = (bf16)v;
        else    Cf[(size_t)m * N + n] = v;
      }
    }
  }
}

// ---------------- Flash attention, 1 wave per (b, h, 16-query tile) ----------------
__global__ __launch_bounds__(32) void attn_wmma_k(
    const bf16* __restrict__ q, const bf16* __restrict__ kb, const bf16* __restrict__ vb,
    bf16* __restrict__ o)
{
  __shared__ __align__(16) bf16 Qs[16 * 64];
  __shared__ __align__(16) bf16 Ks[32 * 64];   // K tile row-major == B^T for Q*K^T
  __shared__ __align__(16) bf16 Vt[64 * 32];   // V^T (d-major) == B^T for P*V
  __shared__ __align__(16) bf16 Ps[16 * 32];   // P tile for layout conversion
  const int lane = threadIdx.x;
  const int hi = lane >> 4, lr = lane & 15;
  const int q0 = blockIdx.x * 16;
  const int bh = blockIdx.y;                   // b*H + h
  const int b  = bh / H_, h = bh % H_;
  const int kvh = h / NREP_;
  const size_t qrow0 = (size_t)b * L_ + q0;

  // Load Q tile [16][64] via async DMA: 128 chunks, 4 per lane
#pragma unroll
  for (int i = 0; i < 4; i++) {
    int c = lane + i * 32;
    int row = c >> 3, cc = c & 7;
    async_copy_b128(&q[(qrow0 + row) * D_ + h * HD_ + cc * 8],
                    &Qs[row * 64 + cc * 8]);
  }
  async_wait0();
  __syncthreads();
  v16bf a0 = frag_ld(&Qs[lr * 64], 0, hi);
  v16bf a1 = frag_ld(&Qs[lr * 64], 32, hi);

  float row_max[8], row_sum[8];
#pragma unroll
  for (int r = 0; r < 8; r++) { row_max[r] = -1.0e30f; row_sum[r] = 0.0f; }
  v8f o0 = {}, o1 = {}, o2 = {}, o3 = {};

  const int jt_max = (q0 + 15) / 32;
  for (int jt = 0; jt <= jt_max; jt++) {
    const int j0 = jt * 32;
    const size_t krow0 = (size_t)b * L_ + j0;
    // stage K tile [32][64] row-major via async DMA: 256 chunks, 8 per lane
#pragma unroll
    for (int i = 0; i < 8; i++) {
      int c = lane + i * 32;
      int row = c >> 3, cc = c & 7;
      async_copy_b128(&kb[(krow0 + row) * KVD_ + kvh * HD_ + cc * 8],
                      &Ks[row * 64 + cc * 8]);
    }
    // stage V tile transposed -> Vt[d][key] (manual: async cannot transpose)
#pragma unroll
    for (int i = 0; i < 8; i++) {
      int c = lane + i * 32;
      int key = c >> 3, cc = c & 7;
      v8bf tv = *(const v8bf*)&vb[(krow0 + key) * KVD_ + kvh * HD_ + cc * 8];
#pragma unroll
      for (int e = 0; e < 8; e++) Vt[(cc * 8 + e) * 32 + key] = tv[e];
    }
    async_wait0();
    __syncthreads();

    // scores: s0 = keys j0..j0+15, s1 = keys j0+16..j0+31 ; HD=64 -> 2 K-steps
    v8f s0 = {}, s1 = {};
    s0 = wmma_bf16(a0, frag_ld(&Ks[lr * 64], 0, hi), s0);
    s0 = wmma_bf16(a1, frag_ld(&Ks[lr * 64], 32, hi), s0);
    s1 = wmma_bf16(a0, frag_ld(&Ks[(16 + lr) * 64], 0, hi), s1);
    s1 = wmma_bf16(a1, frag_ld(&Ks[(16 + lr) * 64], 32, hi), s1);

    // online softmax update (rows m = r + 8*hi live in the lane's 16-lane half)
#pragma unroll
    for (int r = 0; r < 8; r++) {
      int qg = q0 + r + 8 * hi;
      float e0 = s0[r] * 0.125f, e1 = s1[r] * 0.125f;   // HD^-0.5
      if (j0 + lr > qg)      e0 = NEGB;
      if (j0 + 16 + lr > qg) e1 = NEGB;
      float t = fmaxf(e0, e1);
      t = fmaxf(t, __shfl_xor(t, 1, 32));
      t = fmaxf(t, __shfl_xor(t, 2, 32));
      t = fmaxf(t, __shfl_xor(t, 4, 32));
      t = fmaxf(t, __shfl_xor(t, 8, 32));
      float nm = fmaxf(row_max[r], t);
      float alpha = __expf(row_max[r] - nm);
      row_max[r] = nm;
      float p0 = __expf(e0 - nm), p1 = __expf(e1 - nm);
      row_sum[r] = row_sum[r] * alpha + p0 + p1;        // lane-partial sum
      o0[r] *= alpha; o1[r] *= alpha; o2[r] *= alpha; o3[r] *= alpha;
      Ps[(r + 8 * hi) * 32 + lr]      = (bf16)p0;       // C/D layout -> LDS
      Ps[(r + 8 * hi) * 32 + 16 + lr] = (bf16)p1;
    }
    __syncthreads();

    // P (16x32) x V (32x64) in 4 N-chunks of 16
    v16bf pa = frag_ld(&Ps[lr * 32], 0, hi);
    o0 = wmma_bf16(pa, frag_ld(&Vt[(0 * 16 + lr) * 32], 0, hi), o0);
    o1 = wmma_bf16(pa, frag_ld(&Vt[(1 * 16 + lr) * 32], 0, hi), o1);
    o2 = wmma_bf16(pa, frag_ld(&Vt[(2 * 16 + lr) * 32], 0, hi), o2);
    o3 = wmma_bf16(pa, frag_ld(&Vt[(3 * 16 + lr) * 32], 0, hi), o3);
    __syncthreads();
  }

  // finalize: reduce lane-partial sums over the 16-lane half, normalize, store
#pragma unroll
  for (int r = 0; r < 8; r++) {
    float s = row_sum[r];
    s += __shfl_xor(s, 1, 32);
    s += __shfl_xor(s, 2, 32);
    s += __shfl_xor(s, 4, 32);
    s += __shfl_xor(s, 8, 32);
    float inv = 1.0f / s;
    size_t orow = (qrow0 + r + 8 * hi) * D_ + h * HD_;
    o[orow + 0 * 16 + lr] = (bf16)(o0[r] * inv);
    o[orow + 1 * 16 + lr] = (bf16)(o1[r] * inv);
    o[orow + 2 * 16 + lr] = (bf16)(o2[r] * inv);
    o[orow + 3 * 16 + lr] = (bf16)(o3[r] * inv);
  }
}

extern "C" void kernel_launch(void* const* d_in, const int* in_sizes, int n_in,
                              void* d_out, int out_size, void* d_ws, size_t ws_size,
                              hipStream_t stream) {
  (void)in_sizes; (void)n_in; (void)out_size; (void)ws_size;
  const float* x  = (const float*)d_in[0];
  // d_in[1] = mask (unused; causal mask computed analytically)
  const float* Wq = (const float*)d_in[2];
  const float* bq = (const float*)d_in[3];
  const float* Wk = (const float*)d_in[4];
  const float* bk = (const float*)d_in[5];
  const float* Wv = (const float*)d_in[6];
  const float* bv = (const float*)d_in[7];
  const float* Wo = (const float*)d_in[8];
  float* out = (float*)d_out;

  char* ws = (char*)d_ws;
  bf16* xb  = (bf16*)ws; ws += (size_t)BL_ * D_ * 2;
  bf16* wqb = (bf16*)ws; ws += (size_t)D_ * D_ * 2;
  bf16* wkb = (bf16*)ws; ws += (size_t)D_ * KVD_ * 2;
  bf16* wvb = (bf16*)ws; ws += (size_t)D_ * KVD_ * 2;
  bf16* wob = (bf16*)ws; ws += (size_t)D_ * D_ * 2;
  bf16* qb  = (bf16*)ws; ws += (size_t)BL_ * D_ * 2;
  bf16* kbf = (bf16*)ws; ws += (size_t)BL_ * KVD_ * 2;
  bf16* vbf = (bf16*)ws; ws += (size_t)BL_ * KVD_ * 2;
  bf16* ab  = (bf16*)ws; ws += (size_t)BL_ * D_ * 2;

  // fp32 -> bf16 conversions
  f32_to_bf16_k<<<2048, 256, 0, stream>>>(x,  xb,  (size_t)BL_ * D_);
  f32_to_bf16_k<<<2048, 256, 0, stream>>>(Wq, wqb, (size_t)D_ * D_);
  f32_to_bf16_k<<<512,  256, 0, stream>>>(Wk, wkb, (size_t)D_ * KVD_);
  f32_to_bf16_k<<<512,  256, 0, stream>>>(Wv, wvb, (size_t)D_ * KVD_);
  f32_to_bf16_k<<<2048, 256, 0, stream>>>(Wo, wob, (size_t)D_ * D_);

  // projections
  gemm_wmma_k<<<dim3(D_ / GM_NT,  BL_ / GM_MT), 256, 0, stream>>>(
      xb, wqb, bq, qb, nullptr, BL_, D_, D_);
  gemm_wmma_k<<<dim3(KVD_ / GM_NT, BL_ / GM_MT), 256, 0, stream>>>(
      xb, wkb, bk, kbf, nullptr, BL_, KVD_, D_);
  gemm_wmma_k<<<dim3(KVD_ / GM_NT, BL_ / GM_MT), 256, 0, stream>>>(
      xb, wvb, bv, vbf, nullptr, BL_, KVD_, D_);

  // RoPE on Q and K
  rope_k<<<(BL_ * H_ * 32 + 255) / 256, 256, 0, stream>>>(qb, H_, D_, BL_ * H_ * 32);
  rope_k<<<(BL_ * HKV_ * 32 + 255) / 256, 256, 0, stream>>>(kbf, HKV_, KVD_, BL_ * HKV_ * 32);

  // flash attention
  attn_wmma_k<<<dim3(L_ / 16, B_ * H_), 32, 0, stream>>>(qb, kbf, vbf, ab);

  // output projection -> fp32 d_out
  gemm_wmma_k<<<dim3(D_ / GM_NT, BL_ / GM_MT), 256, 0, stream>>>(
      ab, wob, nullptr, nullptr, out, BL_, D_, D_);
}
